// DimeNet_16947940950154
// MI455X (gfx1250) — matrix operand order repaired and checked by reading, not compiled, not run on hardware
//
#include <hip/hip_runtime.h>

typedef __attribute__((ext_vector_type(16))) _Float16 v16h;
typedef __attribute__((ext_vector_type(8)))  _Float16 v8h;
typedef __attribute__((ext_vector_type(8)))  float    v8f;

#define NSW 7
#define NRW 6
#define NBLKW 6
#define NBILW 8
#define MMOL 1024

// ---------------------------------------------------------------------------
// Constant generation (replicates numpy _bessel_zeros / _BNORM / _YPREF)
// ---------------------------------------------------------------------------
__device__ double sph_jn_d(double x, int l) {
    double jm = sin(x) / x;
    if (l == 0) return jm;
    double jc = jm / x - cos(x) / x;
    for (int i = 2; i <= l; ++i) {
        double t = (2.0 * i - 1.0) / x * jc - jm;
        jm = jc; jc = t;
    }
    return jc;
}

__global__ void init_consts_kernel(float* zeros_o, float* bnorm_o, float* ypref_o) {
    if (threadIdx.x != 0 || blockIdx.x != 0) return;
    const double PI = 3.14159265358979323846;
    double Z[NSW][NRW];
    double pts[NRW + NSW - 1];
    for (int k = 0; k < NRW; ++k) Z[0][k] = (k + 1) * PI;
    int npts = NRW + NSW - 1;
    for (int k = 0; k < npts; ++k) pts[k] = (k + 1) * PI;
    for (int l = 1; l < NSW; ++l) {
        double roots[NRW + NSW - 1];
        int nr = npts - 1;
        for (int t = 0; t < nr; ++t) {
            double lo = pts[t], hi = pts[t + 1];
            double flo = sph_jn_d(lo, l);
            for (int it = 0; it < 80; ++it) {
                double mid = 0.5 * (lo + hi);
                double fm = sph_jn_d(mid, l);
                if ((fm >= 0.0) == (flo >= 0.0)) { lo = mid; flo = fm; }
                else hi = mid;
            }
            roots[t] = 0.5 * (lo + hi);
        }
        for (int t = 0; t < nr; ++t) pts[t] = roots[t];
        npts = nr;
        for (int k = 0; k < NRW; ++k) Z[l][k] = roots[k];
    }
    for (int l = 0; l < NSW; ++l)
        for (int k = 0; k < NRW; ++k) {
            zeros_o[l * NRW + k] = (float)Z[l][k];
            bnorm_o[l * NRW + k] = (float)(sqrt(2.0) / fabs(sph_jn_d(Z[l][k], l + 1)));
        }
    for (int l = 0; l < NSW; ++l)
        ypref_o[l] = (float)sqrt((2.0 * l + 1.0) / (4.0 * PI));
}

__global__ void zero_kernel(float* p, long n) {
    long i = blockIdx.x * (long)blockDim.x + threadIdx.x;
    if (i < n) p[i] = 0.0f;
}

// ---------------------------------------------------------------------------
// Weight pre-swizzle: row-major f32 W[k*128+n] (nmat stacked KxN=128 mats)
// -> f16 Wsw[mat][n*K + k]  (n-major, k contiguous = WMMA B-fragment order)
// ---------------------------------------------------------------------------
__global__ void convT_kernel(const float* __restrict__ W, _Float16* __restrict__ Wsw,
                             int K, long total) {
    long idx = blockIdx.x * (long)blockDim.x + threadIdx.x;
    if (idx >= total) return;
    long per = (long)K * 128;
    long mat = idx / per, rem = idx % per;
    long n = rem / K, k = rem % K;
    Wsw[idx] = (_Float16)W[mat * per + k * 128 + n];
}

// plain f32 -> f16 copy (Wbil is already k-contiguous per (i,b) row)
__global__ void conv_kernel(const float* __restrict__ W, _Float16* __restrict__ Wh,
                            long total) {
    long idx = blockIdx.x * (long)blockDim.x + threadIdx.x;
    if (idx < total) Wh[idx] = (_Float16)W[idx];
}

// ---------------------------------------------------------------------------
// Per-edge geometry: dist -> envelope, rbf (E x 6), rbf_s (E x 7 x 6)
// ---------------------------------------------------------------------------
__global__ void edge_kernel(const float* __restrict__ pos,
                            const int* __restrict__ ei, const int* __restrict__ ej,
                            const float* __restrict__ rbf_freq,
                            const float* __restrict__ zeros, const float* __restrict__ bnorm,
                            float* __restrict__ rbf, float* __restrict__ rbf_s, int E) {
    int e = blockIdx.x * blockDim.x + threadIdx.x;
    if (e >= E) return;
    int i = ei[e], j = ej[e];
    float dx = pos[i * 3 + 0] - pos[j * 3 + 0];
    float dy = pos[i * 3 + 1] - pos[j * 3 + 1];
    float dz = pos[i * 3 + 2] - pos[j * 3 + 2];
    float d = sqrtf(dx * dx + dy * dy + dz * dz);
    float x01 = d / 5.0f;
    float x2 = x01 * x01, x4 = x2 * x2;
    float x5 = x4 * x01, x6 = x5 * x01, x7 = x6 * x01;
    float env = 1.0f / x01 - 28.0f * x5 + 48.0f * x6 - 21.0f * x7;
    for (int r = 0; r < NRW; ++r)
        rbf[(long)e * NRW + r] = env * sinf(rbf_freq[r] * x01);
    for (int l = 0; l < NSW; ++l)
        for (int r = 0; r < NRW; ++r) {
            float u = x01 * zeros[l * NRW + r];
            float jm = sinf(u) / u;
            float jv = jm;
            if (l >= 1) {
                float jc = jm / u - cosf(u) / u;
                for (int ll = 2; ll <= l; ++ll) {
                    float t = (2.0f * ll - 1.0f) / u * jc - jm;
                    jm = jc; jc = t;
                }
                jv = jc;
            }
            rbf_s[(long)e * (NSW * NRW) + l * NRW + r] = env * bnorm[l * NRW + r] * jv;
        }
}

// ---------------------------------------------------------------------------
// Per-triplet: angle -> Legendre cbf, sbf (T x 42) = rbf_s[idx_kj] * cbf
// ---------------------------------------------------------------------------
__global__ void triplet_kernel(const float* __restrict__ pos,
                               const int* __restrict__ ti, const int* __restrict__ tj,
                               const int* __restrict__ tk, const int* __restrict__ kjv,
                               const float* __restrict__ ypref,
                               const float* __restrict__ rbf_s,
                               float* __restrict__ sbf, int T) {
    int t = blockIdx.x * blockDim.x + threadIdx.x;
    if (t >= T) return;
    int ia = ti[t], ja = tj[t], ka = tk[t];
    float jx = pos[ja * 3 + 0] - pos[ia * 3 + 0];
    float jy = pos[ja * 3 + 1] - pos[ia * 3 + 1];
    float jz = pos[ja * 3 + 2] - pos[ia * 3 + 2];
    float kx = pos[ka * 3 + 0] - pos[ia * 3 + 0];
    float ky = pos[ka * 3 + 1] - pos[ia * 3 + 1];
    float kz = pos[ka * 3 + 2] - pos[ia * 3 + 2];
    float a = jx * kx + jy * ky + jz * kz;
    float cx = jy * kz - jz * ky;
    float cy = jz * kx - jx * kz;
    float cz = jx * ky - jy * kx;
    float b = sqrtf(cx * cx + cy * cy + cz * cz);
    float angle = atan2f(b, a);
    float ct = cosf(angle);
    float P[NSW];
    P[0] = 1.0f; P[1] = ct;
    for (int l = 2; l < NSW; ++l)
        P[l] = ((2.0f * l - 1.0f) * ct * P[l - 1] - (l - 1.0f) * P[l - 2]) / (float)l;
    int kj = kjv[t];
    for (int l = 0; l < NSW; ++l) {
        float c = ypref[l] * P[l];
        for (int r = 0; r < NRW; ++r)
            sbf[(long)t * (NSW * NRW) + l * NRW + r] =
                rbf_s[(long)kj * (NSW * NRW) + l * NRW + r] * c;
    }
}

// ---------------------------------------------------------------------------
// Naive small-K GEMM (K=6 or 42): out = act(A[MxK] @ B[KxN] + bias)
// ---------------------------------------------------------------------------
__global__ void small_gemm_kernel(const float* __restrict__ A, const float* __restrict__ B,
                                  const float* __restrict__ bias, float* __restrict__ out,
                                  int M, int K, int N, int act) {
    long idx = blockIdx.x * (long)blockDim.x + threadIdx.x;
    if (idx >= (long)M * N) return;
    int m = (int)(idx / N), n = (int)(idx % N);
    float s = 0.0f;
    for (int k = 0; k < K; ++k) s += A[(long)m * K + k] * B[k * N + n];
    if (bias) s += bias[n];
    if (act) s = s / (1.0f + __expf(-s));
    out[idx] = s;
}

// ---------------------------------------------------------------------------
// Build concatenated embedding input xcat[E x 384]
// ---------------------------------------------------------------------------
__global__ void build_xcat_kernel(const float* __restrict__ aemb, const int* __restrict__ z,
                                  const int* __restrict__ ei, const int* __restrict__ ej,
                                  const float* __restrict__ rb, float* __restrict__ xcat, int E) {
    long idx = blockIdx.x * (long)blockDim.x + threadIdx.x;
    if (idx >= (long)E * 384) return;
    int e = (int)(idx / 384), c = (int)(idx % 384);
    float v;
    if (c < 128)      v = aemb[z[ei[e]] * 128 + c];
    else if (c < 256) v = aemb[z[ej[e]] * 128 + (c - 128)];
    else              v = rb[(long)e * 128 + (c - 256)];
    xcat[idx] = v;
}

// ---------------------------------------------------------------------------
// Templated WMMA GEMM, N = 128 (8 waves x 16 cols), M tile = 32 rows.
// f16 operands / f32 accumulate. A double-buffered through LDS (one barrier
// per K-chunk, next chunk prefetched to registers during the WMMAs). B comes
// from a pre-swizzled f16 weight image (fragment = 16 contiguous halfs).
//   A row m:   src = GATHER ? gather[m] : m;   a = A[src*aStride + k]
//              (* rowScale[m*rsStride+rsOff] if SCALE)
//   B frag:    Bsw[bOff + n*bStride + k] .. +15
//   epilogue:  v = acc (+bias[n]) (silu) (*postMul[m,n]) (+resAdd[m,n])
//              SCATTER ? atomicAdd(out[outIdx[m]*128+n], v) : out[m*128+n]=v
// ---------------------------------------------------------------------------
template<int KC, bool GATHER, bool SCALE, bool BIAS, bool ACT,
         bool POST, bool RES, bool SCATTER>
__global__ __launch_bounds__(256)
void wmma_gemm128_kernel(const float* __restrict__ A, int aStride,
                         const int* __restrict__ gather,
                         const float* __restrict__ rowScale, int rsStride, int rsOff,
                         const _Float16* __restrict__ Bsw, int bStride, long bOff,
                         const float* __restrict__ bias,
                         const float* __restrict__ postMul,
                         const float* __restrict__ resAdd,
                         float* __restrict__ out, const int* __restrict__ outIdx,
                         int M) {
    __shared__ _Float16 Asl[2][32 * 32];
    const int tid  = threadIdx.x;
    const int wave = tid >> 5;
    const int lane = tid & 31;
    const int tileM = blockIdx.x * 32;
    const int nBase = wave * 16;

    // per-thread A staging coords: row sr, 4 consecutive k starting at sk
    const int sr = tid >> 3;
    const int sk = (tid & 7) * 4;
    const int sm = tileM + sr;
    const bool sValid = sm < M;
    long srcOff = 0;
    float rs = 1.0f;
    if (sValid) {
        long src = GATHER ? (long)gather[sm] : (long)sm;
        srcOff = src * (long)aStride;
        if (SCALE) rs = rowScale[(long)sm * rsStride + rsOff];
    }

    // B fragment base: 16 contiguous halfs per chunk
    const int ncol = nBase + (lane & 15);
    const int kbase = (lane >> 4) << 4;           // 0 or 16 within chunk
    const _Float16* bBase = Bsw + bOff + (long)ncol * bStride + kbase;

    // A fragment LDS coords (16-bit A 16x32 layout)
    const int mrow = lane & 15;
    const int kh = (lane >> 4) << 4;              // 0 or 16

    v8f cacc[2] = {{}, {}};

    // prologue: stage chunk 0 into buffer 0
    {
        float4 v = make_float4(0.f, 0.f, 0.f, 0.f);
        if (sValid) v = *(const float4*)(A + srcOff + sk);
        if (SCALE) { v.x *= rs; v.y *= rs; v.z *= rs; v.w *= rs; }
        _Float16* dst = &Asl[0][sr * 32 + sk];
        dst[0] = (_Float16)v.x; dst[1] = (_Float16)v.y;
        dst[2] = (_Float16)v.z; dst[3] = (_Float16)v.w;
    }

#pragma unroll
    for (int kc = 0; kc < KC; ++kc) {
        __syncthreads();
        // prefetch next A chunk to registers (overlaps with WMMAs below)
        float4 pf = make_float4(0.f, 0.f, 0.f, 0.f);
        if (kc + 1 < KC && sValid)
            pf = *(const float4*)(A + srcOff + (kc + 1) * 32 + sk);

        // B fragment: two 16B loads of pre-swizzled f16
        const _Float16* bp = bBase + kc * 32;
        v8h blo = *(const v8h*)bp;
        v8h bhi = *(const v8h*)(bp + 8);
        v16h bf = __builtin_shufflevector(blo, bhi,
                    0, 1, 2, 3, 4, 5, 6, 7, 8, 9, 10, 11, 12, 13, 14, 15);

        // A fragments (rows 0-15 and 16-31) from LDS
        const _Float16* as = Asl[kc & 1];
        v16h a0, a1;
#pragma unroll
        for (int v2 = 0; v2 < 4; ++v2) {
            const int ka = kh + 2 * v2;
            const int kb = kh + 8 + 2 * v2;
            a0[2 * v2]         = as[mrow * 32 + ka];
            a0[2 * v2 + 1]     = as[mrow * 32 + ka + 1];
            a0[8 + 2 * v2]     = as[mrow * 32 + kb];
            a0[8 + 2 * v2 + 1] = as[mrow * 32 + kb + 1];
            a1[2 * v2]         = as[(16 + mrow) * 32 + ka];
            a1[2 * v2 + 1]     = as[(16 + mrow) * 32 + ka + 1];
            a1[8 + 2 * v2]     = as[(16 + mrow) * 32 + kb];
            a1[8 + 2 * v2 + 1] = as[(16 + mrow) * 32 + kb + 1];
        }
        cacc[0] = __builtin_amdgcn_wmma_f32_16x16x32_f16(false, a0, false, bf,
                                                         (short)0, cacc[0], false, false);
        cacc[1] = __builtin_amdgcn_wmma_f32_16x16x32_f16(false, a1, false, bf,
                                                         (short)0, cacc[1], false, false);

        // write prefetched chunk to the alternate buffer (its previous readers
        // finished before this iteration's barrier)
        if (kc + 1 < KC) {
            if (SCALE) { pf.x *= rs; pf.y *= rs; pf.z *= rs; pf.w *= rs; }
            _Float16* dst = &Asl[(kc + 1) & 1][sr * 32 + sk];
            dst[0] = (_Float16)pf.x; dst[1] = (_Float16)pf.y;
            dst[2] = (_Float16)pf.z; dst[3] = (_Float16)pf.w;
        }
    }

    // epilogue: C layout VGPR r -> M = r (+8 for lanes 16-31), N = lane % 16
    const int mAdd = (lane >> 4) << 3;
    float bv = 0.0f;
    if (BIAS) bv = bias[ncol];
#pragma unroll
    for (int hh = 0; hh < 2; ++hh) {
#pragma unroll
        for (int r = 0; r < 8; ++r) {
            const int m = tileM + hh * 16 + r + mAdd;
            if (m < M) {
                float v = cacc[hh][r];
                if (BIAS) v += bv;
                if (ACT) v = v / (1.0f + __expf(-v));
                if (POST) v *= postMul[(long)m * 128 + ncol];
                if (RES)  v += resAdd[(long)m * 128 + ncol];
                if (SCATTER) atomicAdd(&out[(long)outIdx[m] * 128 + ncol], v);
                else         out[(long)m * 128 + ncol] = v;
            }
        }
    }
}

// ---------------------------------------------------------------------------
// Molecule accumulation: mol[batch[edge_i[e]]] += x[e]
// ---------------------------------------------------------------------------
__global__ void mol_accum_kernel(const float* __restrict__ x, const int* __restrict__ ei,
                                 const int* __restrict__ batch, float* __restrict__ mol, int E) {
    int e = blockIdx.x;
    if (e >= E) return;
    int c = threadIdx.x;
    atomicAdd(&mol[(long)batch[ei[e]] * 128 + c], x[(long)e * 128 + c]);
}

// ---------------------------------------------------------------------------
// Batch-norm over molecules: one block per column
// ---------------------------------------------------------------------------
__global__ void bn_kernel(const float* __restrict__ mol, const float* __restrict__ gamma,
                          const float* __restrict__ beta, float* __restrict__ hbn) {
    int c = blockIdx.x;
    __shared__ float s1[256], s2[256];
    float a = 0.0f, b = 0.0f;
    for (int m = threadIdx.x; m < MMOL; m += 256) {
        float v = mol[(long)m * 128 + c];
        a += v; b += v * v;
    }
    s1[threadIdx.x] = a; s2[threadIdx.x] = b;
    __syncthreads();
    for (int s = 128; s > 0; s >>= 1) {
        if (threadIdx.x < s) { s1[threadIdx.x] += s1[threadIdx.x + s];
                               s2[threadIdx.x] += s2[threadIdx.x + s]; }
        __syncthreads();
    }
    float mean = s1[0] / (float)MMOL;
    float var  = s2[0] / (float)MMOL - mean * mean;
    float inv  = rsqrtf(var + 1e-5f);
    for (int m = threadIdx.x; m < MMOL; m += 256)
        hbn[(long)m * 128 + c] = (mol[(long)m * 128 + c] - mean) * inv * gamma[c] + beta[c];
}

// ---------------------------------------------------------------------------
// Host orchestration
// ---------------------------------------------------------------------------
extern "C" void kernel_launch(void* const* d_in, const int* in_sizes, int n_in,
                              void* d_out, int out_size, void* d_ws, size_t ws_size,
                              hipStream_t stream) {
    const float* pos       = (const float*)d_in[0];
    const float* rbf_freq  = (const float*)d_in[1];
    const float* atom_emb  = (const float*)d_in[2];
    const float* emb_W_rbf = (const float*)d_in[3];
    const float* emb_b_rbf = (const float*)d_in[4];
    const float* emb_W     = (const float*)d_in[5];
    const float* emb_b     = (const float*)d_in[6];
    const float* blk_W_rbf = (const float*)d_in[7];
    const float* blk_W_sbf = (const float*)d_in[8];
    const float* blk_Wbil  = (const float*)d_in[9];
    const float* blk_Wkj   = (const float*)d_in[10];
    const float* blk_bkj   = (const float*)d_in[11];
    const float* blk_Wji   = (const float*)d_in[12];
    const float* blk_bji   = (const float*)d_in[13];
    const float* blk_resW  = (const float*)d_in[14];
    const float* blk_resb  = (const float*)d_in[15];
    const float* blk_Wskip = (const float*)d_in[16];
    const float* blk_bskip = (const float*)d_in[17];
    const float* bn_gamma  = (const float*)d_in[18];
    const float* bn_beta   = (const float*)d_in[19];
    const float* out_W     = (const float*)d_in[20];
    const float* out_b     = (const float*)d_in[21];
    const int*   z         = (const int*)d_in[22];
    const int*   batch     = (const int*)d_in[23];
    const int*   edge_i    = (const int*)d_in[24];
    const int*   edge_j    = (const int*)d_in[25];
    const int*   idx_t_i   = (const int*)d_in[26];
    const int*   idx_t_j   = (const int*)d_in[27];
    const int*   idx_t_k   = (const int*)d_in[28];
    const int*   idx_kj    = (const int*)d_in[29];
    const int*   idx_ji    = (const int*)d_in[30];

    const int E = in_sizes[24];
    const int T = in_sizes[26];

    // carve workspace (256B aligned regions)
    char* wp = (char*)d_ws;
    auto carve = [&](size_t nfloats) {
        float* p = (float*)wp;
        wp += ((nfloats * sizeof(float) + 255) / 256) * 256;
        return p;
    };
    auto carve16 = [&](size_t nhalf) {
        _Float16* p = (_Float16*)wp;
        wp += ((nhalf * 2 + 255) / 256) * 256;
        return p;
    };
    float* c_zeros = carve(NSW * NRW);
    float* c_bnorm = carve(NSW * NRW);
    float* c_ypref = carve(NSW);
    float* rbf   = carve((size_t)E * NRW);
    float* rbf_s = carve((size_t)E * NSW * NRW);
    float* sbf   = carve((size_t)T * NSW * NRW);
    float* sbfp  = carve((size_t)T * NBILW);
    float* rb    = carve((size_t)E * 128);
    float* xcat  = carve((size_t)E * 384);
    float* xbuf  = carve((size_t)E * 128);
    float* hbuf  = carve((size_t)E * 128);
    float* t1    = carve((size_t)E * 128);
    float* xkj   = carve((size_t)E * 128);
    float* rbfp  = carve((size_t)E * 128);
    float* mol   = carve((size_t)MMOL * 128);
    float* hbn   = carve((size_t)MMOL * 128);
    // pre-swizzled f16 weights
    _Float16* embW_h  = carve16((size_t)384 * 128);
    _Float16* wji_h   = carve16((size_t)NBLKW * 128 * 128);
    _Float16* wkj_h   = carve16((size_t)NBLKW * 128 * 128);
    _Float16* wskip_h = carve16((size_t)NBLKW * 128 * 128);
    _Float16* resw_h  = carve16((size_t)NBLKW * 6 * 128 * 128);
    _Float16* outw_h  = carve16((size_t)128 * 128);
    _Float16* wbil_h  = carve16((size_t)NBLKW * 128 * NBILW * 128);

    const int gE32 = (E + 31) / 32;
    const int gT32 = (T + 31) / 32;

    init_consts_kernel<<<1, 1, 0, stream>>>(c_zeros, c_bnorm, c_ypref);
    zero_kernel<<<(MMOL * 128 + 255) / 256, 256, 0, stream>>>(mol, (long)MMOL * 128);

    // one-time weight swizzle/convert
    {
        long t;
        t = (long)384 * 128;
        convT_kernel<<<(t + 255) / 256, 256, 0, stream>>>(emb_W, embW_h, 384, t);
        t = (long)NBLKW * 128 * 128;
        convT_kernel<<<(t + 255) / 256, 256, 0, stream>>>(blk_Wji, wji_h, 128, t);
        convT_kernel<<<(t + 255) / 256, 256, 0, stream>>>(blk_Wkj, wkj_h, 128, t);
        convT_kernel<<<(t + 255) / 256, 256, 0, stream>>>(blk_Wskip, wskip_h, 128, t);
        t = (long)NBLKW * 6 * 128 * 128;
        convT_kernel<<<(t + 255) / 256, 256, 0, stream>>>(blk_resW, resw_h, 128, t);
        t = (long)128 * 128;
        convT_kernel<<<(t + 255) / 256, 256, 0, stream>>>(out_W, outw_h, 128, t);
        t = (long)NBLKW * 128 * NBILW * 128;
        conv_kernel<<<(t + 255) / 256, 256, 0, stream>>>(blk_Wbil, wbil_h, t);
    }

    edge_kernel<<<(E + 127) / 128, 128, 0, stream>>>(pos, edge_i, edge_j, rbf_freq,
                                                     c_zeros, c_bnorm, rbf, rbf_s, E);
    triplet_kernel<<<(T + 127) / 128, 128, 0, stream>>>(pos, idx_t_i, idx_t_j, idx_t_k,
                                                        idx_kj, c_ypref, rbf_s, sbf, T);

    // rb = silu(rbf @ emb_W_rbf + emb_b_rbf)    (K=6)
    small_gemm_kernel<<<((long)E * 128 + 255) / 256, 256, 0, stream>>>(
        rbf, emb_W_rbf, emb_b_rbf, rb, E, NRW, 128, 1);
    // xcat = [aemb[z_i] | aemb[z_j] | rb]
    build_xcat_kernel<<<((long)E * 384 + 255) / 256, 256, 0, stream>>>(
        atom_emb, z, edge_i, edge_j, rb, xcat, E);
    // x = silu(xcat @ emb_W + emb_b)            (K=384, WMMA)
    wmma_gemm128_kernel<12, false, false, true, true, false, false, false>
        <<<gE32, 256, 0, stream>>>(
        xcat, 384, nullptr, nullptr, 0, 0,
        embW_h, 384, 0, emb_b, nullptr, nullptr, xbuf, nullptr, E);

    for (int blk = 0; blk < NBLKW; ++blk) {
        const float* bji   = blk_bji   + (long)blk * 128;
        const float* bkj   = blk_bkj   + (long)blk * 128;
        const float* bskip = blk_bskip + (long)blk * 128;
        const long   wOff  = (long)blk * 128 * 128;

        // rbf_p = rbf @ blk_W_rbf[blk]          (K=6)
        small_gemm_kernel<<<((long)E * 128 + 255) / 256, 256, 0, stream>>>(
            rbf, blk_W_rbf + (long)blk * NRW * 128, nullptr, rbfp, E, NRW, 128, 0);
        // sbf_p = sbf @ blk_W_sbf[blk]          (K=42, N=8)
        small_gemm_kernel<<<((long)T * NBILW + 255) / 256, 256, 0, stream>>>(
            sbf, blk_W_sbf + (long)blk * NSW * NRW * NBILW, nullptr, sbfp,
            T, NSW * NRW, NBILW, 0);

        // h := x_ji = silu(x @ Wji + bji)       (WMMA, store into hbuf)
        wmma_gemm128_kernel<4, false, false, true, true, false, false, false>
            <<<gE32, 256, 0, stream>>>(
            xbuf, 128, nullptr, nullptr, 0, 0,
            wji_h, 128, wOff, bji, nullptr, nullptr, hbuf, nullptr, E);
        // x_kj = silu(x @ Wkj + bkj) * rbf_p    (WMMA)
        wmma_gemm128_kernel<4, false, false, true, true, true, false, false>
            <<<gE32, 256, 0, stream>>>(
            xbuf, 128, nullptr, nullptr, 0, 0,
            wkj_h, 128, wOff, bkj, rbfp, nullptr, xkj, nullptr, E);

        // einsum 'tb,th,ibh->ti' as 8 scaled gather-GEMMs, atomic scatter over idx_ji
        for (int b = 0; b < NBILW; ++b) {
            long bOff = (long)blk * 128 * NBILW * 128 + (long)b * 128;
            wmma_gemm128_kernel<4, true, true, false, false, false, false, true>
                <<<gT32, 256, 0, stream>>>(
                xkj, 128, idx_kj, sbfp, NBILW, b,
                wbil_h, NBILW * 128, bOff,
                nullptr, nullptr, nullptr, hbuf, idx_ji, T);
        }

        // residual layer 0 on h (hbuf)
        {
            long w1 = ((long)(blk * 3 + 0) * 2 + 0) * 128 * 128;
            long w2 = ((long)(blk * 3 + 0) * 2 + 1) * 128 * 128;
            const float* b1 = blk_resb + ((long)(blk * 3 + 0) * 2 + 0) * 128;
            const float* b2 = blk_resb + ((long)(blk * 3 + 0) * 2 + 1) * 128;
            wmma_gemm128_kernel<4, false, false, true, true, false, false, false>
                <<<gE32, 256, 0, stream>>>(
                hbuf, 128, nullptr, nullptr, 0, 0,
                resw_h, 128, w1, b1, nullptr, nullptr, t1, nullptr, E);
            wmma_gemm128_kernel<4, false, false, true, true, false, true, false>
                <<<gE32, 256, 0, stream>>>(
                t1, 128, nullptr, nullptr, 0, 0,
                resw_h, 128, w2, b2, nullptr, hbuf, hbuf, nullptr, E);
        }
        // x = silu(h @ Wskip + bskip) + x
        wmma_gemm128_kernel<4, false, false, true, true, false, true, false>
            <<<gE32, 256, 0, stream>>>(
            hbuf, 128, nullptr, nullptr, 0, 0,
            wskip_h, 128, wOff, bskip, nullptr, xbuf, xbuf, nullptr, E);
        // residual layers 1,2 on x
        for (int r = 1; r < 3; ++r) {
            long w1 = ((long)(blk * 3 + r) * 2 + 0) * 128 * 128;
            long w2 = ((long)(blk * 3 + r) * 2 + 1) * 128 * 128;
            const float* b1 = blk_resb + ((long)(blk * 3 + r) * 2 + 0) * 128;
            const float* b2 = blk_resb + ((long)(blk * 3 + r) * 2 + 1) * 128;
            wmma_gemm128_kernel<4, false, false, true, true, false, false, false>
                <<<gE32, 256, 0, stream>>>(
                xbuf, 128, nullptr, nullptr, 0, 0,
                resw_h, 128, w1, b1, nullptr, nullptr, t1, nullptr, E);
            wmma_gemm128_kernel<4, false, false, true, true, false, true, false>
                <<<gE32, 256, 0, stream>>>(
                t1, 128, nullptr, nullptr, 0, 0,
                resw_h, 128, w2, b2, nullptr, xbuf, xbuf, nullptr, E);
        }
        // mol_acc += segsum(segsum(x, edge_i), batch)
        mol_accum_kernel<<<E, 128, 0, stream>>>(xbuf, edge_i, batch, mol, E);
    }

    // batch-norm over molecules, then final projection (WMMA)
    bn_kernel<<<128, 256, 0, stream>>>(mol, bn_gamma, bn_beta, hbn);
    wmma_gemm128_kernel<4, false, false, true, false, false, false, false>
        <<<(MMOL + 31) / 32, 256, 0, stream>>>(
        hbn, 128, nullptr, nullptr, 0, 0,
        outw_h, 128, 0, out_b, nullptr, nullptr, (float*)d_out, nullptr, MMOL);
}
